// NetVLAD_58239756533954
// MI455X (gfx1250) — compile-verified, matrix operand order checked by reading, not run on hardware
//
#include <hip/hip_runtime.h>
#include <math.h>

// ---- problem constants (from reference setup_inputs) ----
#define Bb   32
#define Cc   128
#define Nn   16384      // H*W = 128*128
#define Kc   64
#define FEPS 1e-12f

typedef __attribute__((ext_vector_type(16))) __bf16 v16bf;
typedef __attribute__((ext_vector_type(8)))  __bf16 v8bf;
typedef __attribute__((ext_vector_type(8)))  float  v8f;

// ISA bf16 A/B fragment mapping for 16x16x32 WMMA:
//   reduction index k (0..31), row/col index mn (0..15)
//   hi = (k>>3)&1 ; i = (k&7) + ((k>>4)<<3) ; lane = mn + 16*hi
__device__ __forceinline__ void frag_pos(int k, int mn, int& lane, int& i) {
    int hi = (k >> 3) & 1;
    i    = (k & 7) + ((k >> 4) << 3);
    lane = mn + (hi << 4);
}

// Load one lane's 16-half fragment (32 contiguous bytes) as 2x ds_load_b128
__device__ __forceinline__ v16bf ld_frag(const __bf16* p) {
    v8bf lo = *(const v8bf*)p;
    v8bf hi = *(const v8bf*)(p + 8);
    return __builtin_shufflevector(lo, hi, 0, 1, 2, 3, 4, 5, 6, 7,
                                           8, 9, 10, 11, 12, 13, 14, 15);
}

// ------------------------------------------------------------------
// 0) zero vlad output region + per-batch global-norm accumulators
// ------------------------------------------------------------------
__global__ void k_zero(float* __restrict__ vlad, float* __restrict__ gnorm) {
    int i = blockIdx.x * blockDim.x + threadIdx.x;
    if (i < Bb * Kc * Cc) vlad[i] = 0.0f;
    if (i < Bb)           gnorm[i] = 0.0f;
}

// ------------------------------------------------------------------
// 1) rnorm[b,n] = 1 / max(||x[b,:,n]||_2, eps)   (coalesced over n)
// ------------------------------------------------------------------
__global__ __launch_bounds__(256) void k_rnorm(const float* __restrict__ x,
                                               float* __restrict__ rnorm) {
    int i = blockIdx.x * blockDim.x + threadIdx.x;   // over B*N
    if (i >= Bb * Nn) return;
    int b = i / Nn, n = i % Nn;
    const float* p = x + (size_t)b * Cc * Nn + n;
    float s = 0.0f;
#pragma unroll 8
    for (int c = 0; c < Cc; ++c) {
        float v = p[(size_t)c * Nn];
        s += v * v;
    }
    rnorm[i] = 1.0f / fmaxf(sqrtf(s), FEPS);
}

// ------------------------------------------------------------------
// 2) logits GEMM: soft_region[b,k,n] = (W @ x)[k,n] * rnorm[b,n] + bias[k]
//    64(K) x 32(n) per block, 8 waves -> one 16x16 tile each,
//    C-loop = 4 slabs of 32; LDS held in fragment-major layout so each
//    operand is two ds_load_b128 per lane.
// ------------------------------------------------------------------
__global__ __launch_bounds__(256) void k_logits(const float* __restrict__ x,
                                                const float* __restrict__ conv_w,
                                                const float* __restrict__ conv_b,
                                                const float* __restrict__ rnorm,
                                                float* __restrict__ logits) {
    __shared__ __bf16 Wf[4][4][32][16];   // [rowTile][slab][lane][i] : 16 KB
    __shared__ __bf16 Xf[4][2][32][16];   // [slab][colTile][lane][i] :  8 KB
    __shared__ float  Os[Kc][33];         // output transpose buffer : 8.25 KB

    const int b     = blockIdx.y;
    const int ncol0 = blockIdx.x * 32;
    const int tid   = threadIdx.x;

    // stage conv_w (64x128) into fragment-major bf16
    for (int idx = tid; idx < Kc * Cc; idx += 256) {
        int k = idx >> 7, c = idx & 127;
        int lane, i;
        frag_pos(c & 31, k & 15, lane, i);
        Wf[k >> 4][c >> 5][lane][i] = (__bf16)conv_w[idx];
    }
    // stage x tile (128 x 32) into fragment-major bf16 (coalesced over n)
    for (int idx = tid; idx < Cc * 32; idx += 256) {
        int j = idx & 31, c = idx >> 5;
        float v = x[(size_t)b * Cc * Nn + (size_t)c * Nn + ncol0 + j];
        int lane, i;
        frag_pos(c & 31, j & 15, lane, i);
        Xf[c >> 5][j >> 4][lane][i] = (__bf16)v;
    }
    __syncthreads();

    const int wave = tid >> 5, lane = tid & 31;
    const int r  = wave & 3;           // K row tile: rows 16r..16r+15
    const int ct = wave >> 2;          // n col tile: cols 16ct..16ct+15
    const int m  = lane & 15;
    const int hi = lane >> 4;

    v8f acc = {};
#pragma unroll
    for (int slab = 0; slab < 4; ++slab) {
        v16bf a  = ld_frag(&Wf[r][slab][lane][0]);
        v16bf bf = ld_frag(&Xf[slab][ct][lane][0]);
        acc = __builtin_amdgcn_wmma_f32_16x16x32_bf16(false, a, false, bf,
                                                      (short)0, acc, false, false);
    }

    // fold column normalization + bias; transpose through LDS for coalesced stores
    const int n_l = 16 * ct + m;
    const float rs = rnorm[b * Nn + ncol0 + n_l];
#pragma unroll
    for (int e = 0; e < 8; ++e) {
        int kg = 16 * r + e + (hi ? 8 : 0);            // C/D layout
        Os[kg][n_l] = acc[e] * rs + conv_b[kg];
    }
    __syncthreads();
    for (int idx = tid; idx < Kc * 32; idx += 256) {
        int k = idx >> 5, j = idx & 31;
        logits[((size_t)(b * Kc + k)) * Nn + ncol0 + j] = Os[k][j];
    }
}

// ------------------------------------------------------------------
// 3) softmax over the SPATIAL axis (N=16384) per (b,k) row, in place
// ------------------------------------------------------------------
__global__ __launch_bounds__(256) void k_softmax(float* __restrict__ soft) {
    __shared__ float red[256];
    float* p = soft + (size_t)blockIdx.x * Nn;   // row = b*K + k
    const int tid = threadIdx.x;

    float mx = -3.402823466e38f;
    for (int n = tid; n < Nn; n += 256) mx = fmaxf(mx, p[n]);
    red[tid] = mx; __syncthreads();
    for (int s = 128; s > 0; s >>= 1) {
        if (tid < s) red[tid] = fmaxf(red[tid], red[tid + s]);
        __syncthreads();
    }
    mx = red[0]; __syncthreads();

    float sum = 0.0f;
    for (int n = tid; n < Nn; n += 256) {
        float e = __expf(p[n] - mx);
        p[n] = e;
        sum += e;
    }
    red[tid] = sum; __syncthreads();
    for (int s = 128; s > 0; s >>= 1) {
        if (tid < s) red[tid] += red[tid + s];
        __syncthreads();
    }
    const float inv = 1.0f / red[0];
    for (int n = tid; n < Nn; n += 256) p[n] *= inv;
}

// ------------------------------------------------------------------
// 4) VLAD GEMM: vlad[b,k,c] += sum_n (soft[b,k,n]*rnorm[b,n]) * x[b,c,n]
//    split-N reduction; 64(K) x 128(C) per block, 8 waves x 4 tiles each;
//    fragment-major LDS; f32 atomic adds for the partials
// ------------------------------------------------------------------
__global__ __launch_bounds__(256) void k_vlad(const float* __restrict__ x,
                                              const float* __restrict__ soft,
                                              const float* __restrict__ rnorm,
                                              float* __restrict__ vlad) {
    __shared__ __bf16 Af[4][32][16];    // [rowTile][lane][i] : 4 KB
    __shared__ __bf16 Bf[8][32][16];    // [colTile][lane][i] : 8 KB

    const int CHUNK = 1024;
    const int b      = blockIdx.y;
    const int nbase0 = blockIdx.x * CHUNK;
    const int tid    = threadIdx.x;

    const int wave = tid >> 5, lane = tid & 31;
    const int r  = wave & 3;            // K row tile
    const int cg = wave >> 2;           // C col group: tiles 4cg .. 4cg+3
    const int m  = lane & 15;
    const int hi = lane >> 4;

    v8f a0 = {}, a1 = {}, a2 = {}, a3 = {};

    for (int nb = nbase0; nb < nbase0 + CHUNK; nb += 32) {
        __syncthreads();
        // A tile: soft * rnorm (column normalization of x folded into soft)
        for (int idx = tid; idx < Kc * 32; idx += 256) {
            int j = idx & 31, k = idx >> 5;
            float s = soft[((size_t)(b * Kc + k)) * Nn + nb + j]
                    * rnorm[b * Nn + nb + j];
            int ln, ii;
            frag_pos(j, k & 15, ln, ii);
            Af[k >> 4][ln][ii] = (__bf16)s;
        }
        // B tile: xT[n][c] (coalesced global reads over n)
        for (int idx = tid; idx < 32 * Cc; idx += 256) {
            int j = idx & 31, c = idx >> 5;
            float v = x[(size_t)b * Cc * Nn + (size_t)c * Nn + nb + j];
            int ln, ii;
            frag_pos(j, c & 15, ln, ii);
            Bf[c >> 4][ln][ii] = (__bf16)v;
        }
        __syncthreads();

        v16bf a  = ld_frag(&Af[r][lane][0]);
        v16bf b0 = ld_frag(&Bf[4 * cg + 0][lane][0]);
        v16bf b1 = ld_frag(&Bf[4 * cg + 1][lane][0]);
        v16bf b2 = ld_frag(&Bf[4 * cg + 2][lane][0]);
        v16bf b3 = ld_frag(&Bf[4 * cg + 3][lane][0]);
        a0 = __builtin_amdgcn_wmma_f32_16x16x32_bf16(false, a, false, b0, (short)0, a0, false, false);
        a1 = __builtin_amdgcn_wmma_f32_16x16x32_bf16(false, a, false, b1, (short)0, a1, false, false);
        a2 = __builtin_amdgcn_wmma_f32_16x16x32_bf16(false, a, false, b2, (short)0, a2, false, false);
        a3 = __builtin_amdgcn_wmma_f32_16x16x32_bf16(false, a, false, b3, (short)0, a3, false, false);
    }

    // split-N partial reduction into global vlad
#pragma unroll
    for (int e = 0; e < 8; ++e) {
        int kg = 16 * r + e + (hi ? 8 : 0);
        float* dst = &vlad[((size_t)(b * Kc + kg)) * Cc + 64 * cg + m];
        atomicAdd(dst +  0, a0[e]);
        atomicAdd(dst + 16, a1[e]);
        atomicAdd(dst + 32, a2[e]);
        atomicAdd(dst + 48, a3[e]);
    }
}

// ------------------------------------------------------------------
// 5) cluster subtract (soft_sum == 1 exactly) + intra-normalize per (b,k);
//    accumulate per-batch global sum of squares
// ------------------------------------------------------------------
__global__ __launch_bounds__(128) void k_intra(const float* __restrict__ clusters,
                                               float* __restrict__ vlad,
                                               float* __restrict__ gnorm) {
    __shared__ float red[128];
    const int row = blockIdx.x;         // b*K + k
    const int k   = row & (Kc - 1);
    const int c   = threadIdx.x;

    float v = vlad[(size_t)row * Cc + c] - clusters[k * Cc + c];
    red[c] = v * v; __syncthreads();
    for (int s = 64; s > 0; s >>= 1) {
        if (c < s) red[c] += red[c + s];
        __syncthreads();
    }
    const float ss  = red[0];
    const float inv = 1.0f / fmaxf(sqrtf(ss), FEPS);
    vlad[(size_t)row * Cc + c] = v * inv;
    if (c == 0) atomicAdd(&gnorm[row / Kc], ss * inv * inv);
}

// ------------------------------------------------------------------
// 6) global L2 normalize per batch over K*C
// ------------------------------------------------------------------
__global__ void k_gscale(float* __restrict__ vlad, const float* __restrict__ gnorm) {
    int i = blockIdx.x * blockDim.x + threadIdx.x;   // over B*K*C
    if (i >= Bb * Kc * Cc) return;
    int b = i / (Kc * Cc);
    vlad[i] *= 1.0f / fmaxf(sqrtf(gnorm[b]), FEPS);
}

// ------------------------------------------------------------------
extern "C" void kernel_launch(void* const* d_in, const int* in_sizes, int n_in,
                              void* d_out, int out_size, void* d_ws, size_t ws_size,
                              hipStream_t stream) {
    (void)in_sizes; (void)n_in; (void)out_size; (void)ws_size;
    const float* x        = (const float*)d_in[0];   // (B,C,H,W)
    const float* clusters = (const float*)d_in[1];   // (K,C)
    const float* conv_w   = (const float*)d_in[2];   // (K,C)
    const float* conv_b   = (const float*)d_in[3];   // (K,)

    float* vlad = (float*)d_out;                     // (B, K*C)
    float* soft = (float*)d_out + (size_t)Bb * Kc * Cc;   // (B,K,H,W)

    float* rnorm = (float*)d_ws;                     // B*N floats (2 MB)
    float* gnorm = rnorm + (size_t)Bb * Nn;          // B floats

    k_zero   <<<(Bb * Kc * Cc + 255) / 256, 256, 0, stream>>>(vlad, gnorm);
    k_rnorm  <<<(Bb * Nn) / 256, 256, 0, stream>>>(x, rnorm);
    k_logits <<<dim3(Nn / 32, Bb), 256, 0, stream>>>(x, conv_w, conv_b, rnorm, soft);
    k_softmax<<<Bb * Kc, 256, 0, stream>>>(soft);
    k_vlad   <<<dim3(Nn / 1024, Bb), 256, 0, stream>>>(x, soft, rnorm, vlad);
    k_intra  <<<Bb * Kc, 128, 0, stream>>>(clusters, vlad, gnorm);
    k_gscale <<<(Bb * Kc * Cc + 255) / 256, 256, 0, stream>>>(vlad, gnorm);
}
